// S4ConditionalModel_10986526343832
// MI455X (gfx1250) — matrix operand order repaired and checked by reading, not compiled, not run on hardware
//
#include <hip/hip_runtime.h>
#include <hip/hip_bf16.h>

#define Bdim   16
#define Ldim   16384
#define Hdim   64
#define NCc    8
#define DEPTHc 6
#define CONDc  32
#define CHUNK  128
#define NCHUNK (Ldim / CHUNK)
#define LTILE  32
#define LRANGE 512   // l-range per mix workgroup

typedef __attribute__((ext_vector_type(16))) __bf16 v16bf;
typedef __attribute__((ext_vector_type(8)))  float  v8f;

// Branchless GELU (tanh form) from native gfx1250 transcendentals:
// one v_exp_f32 + one v_rcp_f32, clamp instead of branches.
__device__ __forceinline__ float gelu_fast(float v) {
    float t = v * (0.7978845608028654f + 0.035677408136300125f * v * v);
    t = fminf(fmaxf(t, -15.f), 15.f);
    float e  = __builtin_amdgcn_exp2f(t * 2.8853900817779268f);  // exp(2t)
    float th = 1.0f - 2.0f * __builtin_amdgcn_rcpf(e + 1.0f);    // tanh(t)
    return 0.5f * v * (1.0f + th);
}

// ---------------------------------------------------------------------------
// Conditioning MLP + FiLM precompute:  gb[d][b][0..2H) = c[b] @ film_w[d] + film_b[d]
// ---------------------------------------------------------------------------
__global__ void cond_film_kernel(const float* __restrict__ params,
                                 const float* __restrict__ w1, const float* __restrict__ b1,
                                 const float* __restrict__ w2, const float* __restrict__ b2,
                                 const float* __restrict__ w3, const float* __restrict__ b3,
                                 const float* __restrict__ film_w, const float* __restrict__ film_b,
                                 float* __restrict__ gbout) {
    __shared__ float c[Bdim][CONDc];
    int tid = threadIdx.x;
    if (tid < Bdim) {
        float p0 = params[tid * 2 + 0], p1 = params[tid * 2 + 1];
        float t1[16];
        #pragma unroll
        for (int j = 0; j < 16; ++j)
            t1[j] = fmaxf(0.f, p0 * w1[j] + p1 * w1[16 + j] + b1[j]);
        float t2[32];
        #pragma unroll
        for (int j = 0; j < 32; ++j) {
            float a = b2[j];
            #pragma unroll
            for (int k = 0; k < 16; ++k) a = fmaf(t1[k], w2[k * 32 + j], a);
            t2[j] = fmaxf(0.f, a);
        }
        #pragma unroll
        for (int j = 0; j < 32; ++j) {
            float a = b3[j];
            #pragma unroll
            for (int k = 0; k < 32; ++k) a = fmaf(t2[k], w3[k * 32 + j], a);
            c[tid][j] = fmaxf(0.f, a);
        }
    }
    __syncthreads();
    for (int e = tid; e < DEPTHc * Bdim * 2 * Hdim; e += blockDim.x) {
        int j = e % (2 * Hdim);
        int r = e / (2 * Hdim);
        int b = r % Bdim;
        int d = r / Bdim;
        float a = film_b[d * 2 * Hdim + j];
        #pragma unroll
        for (int k = 0; k < CONDc; ++k)
            a = fmaf(c[b][k], film_w[((size_t)d * CONDc + k) * 2 * Hdim + j], a);
        gbout[e] = a;
    }
}

// ---------------------------------------------------------------------------
// SSM discretization: per (d,h,n) store {λre, λim, λ^CHUNK re, λ^CHUNK im, 2C̃re, 2C̃im, 0, 0}
// ---------------------------------------------------------------------------
__global__ void ssm_prep_kernel(const float* __restrict__ log_dt,
                                const float* __restrict__ log_A_real,
                                const float* __restrict__ A_imag,
                                const float* __restrict__ C_re,
                                const float* __restrict__ C_im,
                                float* __restrict__ out) {
    int idx = blockIdx.x * blockDim.x + threadIdx.x;
    if (idx >= DEPTHc * Hdim * NCc) return;
    int dh = idx / NCc;
    float dt  = expf(log_dt[dh]);
    float are = -expf(log_A_real[idx]);
    float aim = A_imag[idx];
    float dre = are * dt, dim = aim * dt;
    float er  = expf(dre);
    float lre = er * cosf(dim), lim = er * sinf(dim);
    // C~ = C * (λ - 1) / A
    float cr = C_re[idx], ci = C_im[idx];
    float nr = lre - 1.0f, ni = lim;
    float tr = cr * nr - ci * ni;
    float ti = cr * ni + ci * nr;
    float den = 1.0f / (are * are + aim * aim);
    float ctre = (tr * are + ti * aim) * den;
    float ctim = (ti * are - tr * aim) * den;
    // λ^CHUNK computed exactly from dtA
    float pr  = expf(dre * (float)CHUNK);
    float ang = dim * (float)CHUNK;
    float* o = out + (size_t)idx * 8;
    o[0] = lre; o[1] = lim;
    o[2] = pr * cosf(ang); o[3] = pr * sinf(ang);
    o[4] = 2.0f * ctre; o[5] = 2.0f * ctim;
    o[6] = 0.f; o[7] = 0.f;
}

// ---------------------------------------------------------------------------
// expand: h[b][l][h] = x[b][l] * ew[h] + eb[h]   (layout (B,L,H), h fastest)
// ---------------------------------------------------------------------------
__global__ void expand_kernel(const float* __restrict__ x,
                              const float* __restrict__ ew,
                              const float* __restrict__ eb,
                              float* __restrict__ hout) {
    size_t idx = (size_t)blockIdx.x * blockDim.x + threadIdx.x;
    int h  = (int)(idx & (Hdim - 1));
    size_t bl = idx >> 6;
    hout[idx] = fmaf(x[bl], ew[h], eb[h]);
}

// ---------------------------------------------------------------------------
// Channel mix + GELU via bf16 WMMA:  u[b,l,o] = gelu( Σ_h hin[b,l,h] * W[o,h] + bias[o] )
// A = activation tile (16 l-rows × 32 h), B = Wᵀ tile (32 h × 16 o), f32 accum.
// ---------------------------------------------------------------------------
__global__ __launch_bounds__(256) void mix_gelu_wmma_kernel(
        const float* __restrict__ hin, const float* __restrict__ W,
        const float* __restrict__ bias, float* __restrict__ uout) {
    int b = blockIdx.y;
    int l0base = blockIdx.x * LRANGE;
    const float* hb = hin + (size_t)b * Ldim * Hdim;
    float*       ub = uout + (size_t)b * Ldim * Hdim;

    __shared__ __bf16 sW[Hdim * Hdim];   // sW[o*64 + h]
    __shared__ __bf16 sA[LTILE * Hdim];  // sA[(l - l0)*64 + h]

    int tid  = threadIdx.x;
    for (int e = tid; e < Hdim * Hdim; e += 256) sW[e] = (__bf16)W[e];

    int wave = tid >> 5, lane = tid & 31;
    int lsub = (wave & 1) * 16;        // l sub-tile
    int o0   = (wave >> 1) * 16;       // o tile
    int m    = lane & 15;
    int hiA  = (lane >= 16) ? 8 : 0;   // unused: A fragment also uses hiB
    int hiB  = (lane >= 16) ? 16 : 0;  // B-fragment lane-half k offset
    float biasv = bias[o0 + m];

    for (int it = 0; it < LRANGE / LTILE; ++it) {
        int l0 = l0base + it * LTILE;
        __syncthreads();   // sW ready (it==0) / previous sA reads done
        for (int e = tid; e < LTILE * Hdim; e += 256)
            sA[e] = (__bf16)hb[((size_t)(l0 + (e >> 6))) * Hdim + (e & 63)];
        if (it + 1 < LRANGE / LTILE && tid < LTILE)
            __builtin_prefetch(hb + (size_t)(l0 + LTILE + tid) * Hdim, 0, 0);
        __syncthreads();

        v8f acc = {0.f, 0.f, 0.f, 0.f, 0.f, 0.f, 0.f, 0.f};
        #pragma unroll
        for (int kb = 0; kb < Hdim; kb += 32) {
            v16bf afrag, bfrag;
            #pragma unroll
            for (int i = 0; i < 16; ++i) {
                int ka = kb + i + hiB;                        // contiguous k = 16*half + i, matches B
                int kbk = kb + i + hiB;                       // contiguous k = 16*half + i, same order as A
                afrag[i] = sA[(lsub + m) * Hdim + ka];
                bfrag[i] = sW[(o0 + m) * Hdim + kbk];
            }
            acc = __builtin_amdgcn_wmma_f32_16x16x32_bf16(
                      false, afrag, false, bfrag, (short)0, acc, false, false);
        }
        // C/D layout: VGPR j -> row l0+lsub+j (+8 for lanes 16..31), col o0+(lane&15)
        int rowoff = l0 + lsub + ((lane >= 16) ? 8 : 0);
        int col = o0 + m;
        #pragma unroll
        for (int j = 0; j < 8; ++j) {
            float v = acc[j] + biasv;
            ub[(size_t)(rowoff + j) * Hdim + col] = gelu_fast(v);
        }
    }
}

// ---------------------------------------------------------------------------
// S4D chunked scan, phase 1: per-(b,chunk,h) local end-state with zero init
// ---------------------------------------------------------------------------
__global__ __launch_bounds__(64) void scan_phase1_kernel(
        const float* __restrict__ u, const float* __restrict__ ssm,
        float* __restrict__ loc) {
    int b = blockIdx.y, ch = blockIdx.x, h = threadIdx.x;
    const float* p = ssm + (size_t)h * NCc * 8;
    float lre[NCc], lim[NCc], sre[NCc], sim[NCc];
    #pragma unroll
    for (int n = 0; n < NCc; ++n) {
        const float4 q = *(const float4*)(p + n * 8);
        lre[n] = q.x; lim[n] = q.y; sre[n] = 0.f; sim[n] = 0.f;
    }
    const float* ubp = u + ((size_t)b * Ldim + (size_t)ch * CHUNK) * Hdim + h;
    #pragma unroll 4
    for (int l = 0; l < CHUNK; ++l) {
        float uv = ubp[(size_t)l * Hdim];
        #pragma unroll
        for (int n = 0; n < NCc; ++n) {
            float r = sre[n], ii = sim[n];
            sre[n] = fmaf(lre[n], r, fmaf(-lim[n], ii, uv));
            sim[n] = fmaf(lre[n], ii, lim[n] * r);
        }
    }
    float* o = loc + (((size_t)b * NCHUNK + ch) * Hdim + h) * 16;
    #pragma unroll
    for (int n = 0; n < NCc; ++n) { o[2 * n] = sre[n]; o[2 * n + 1] = sim[n]; }
}

// ---------------------------------------------------------------------------
// phase 2: scan the 128 chunk boundaries per (b,h):  carry[j] = λ^C·carry[j-1] + loc[j-1]
// ---------------------------------------------------------------------------
__global__ void scan_phase2_kernel(const float* __restrict__ ssm,
                                   const float* __restrict__ loc,
                                   float* __restrict__ carry) {
    int idx = blockIdx.x * blockDim.x + threadIdx.x;
    if (idx >= Bdim * Hdim) return;
    int b = idx / Hdim, h = idx % Hdim;
    const float* p = ssm + (size_t)h * NCc * 8;
    float lpre[NCc], lpim[NCc], cre[NCc], cim[NCc];
    #pragma unroll
    for (int n = 0; n < NCc; ++n) {
        lpre[n] = p[n * 8 + 2]; lpim[n] = p[n * 8 + 3];
        cre[n] = 0.f; cim[n] = 0.f;
    }
    for (int ch = 0; ch < NCHUNK; ++ch) {
        size_t off = (((size_t)b * NCHUNK + ch) * Hdim + h) * 16;
        float* co = carry + off;
        const float* li = loc + off;
        #pragma unroll
        for (int n = 0; n < NCc; ++n) {
            co[2 * n] = cre[n]; co[2 * n + 1] = cim[n];
            float r = cre[n], ii = cim[n];
            cre[n] = fmaf(lpre[n], r, fmaf(-lpim[n], ii, li[2 * n]));
            cim[n] = fmaf(lpre[n], ii, fmaf(lpim[n], r, li[2 * n + 1]));
        }
    }
}

// ---------------------------------------------------------------------------
// phase 3: recompute with correct carry-in; fuse D-skip, FiLM, GELU, residual.
// hio is read (x_in) and overwritten (new h) elementwise by the same thread.
// ---------------------------------------------------------------------------
__global__ __launch_bounds__(64) void scan_phase3_kernel(
        const float* __restrict__ u, float* __restrict__ hio,
        const float* __restrict__ ssm, const float* __restrict__ Dvec,
        const float* __restrict__ gb, const float* __restrict__ resw,
        const float* __restrict__ carry) {
    int b = blockIdx.y, ch = blockIdx.x, h = threadIdx.x;
    const float* p = ssm + (size_t)h * NCc * 8;
    const float* ci = carry + (((size_t)b * NCHUNK + ch) * Hdim + h) * 16;
    float lre[NCc], lim[NCc], ctre[NCc], ctim[NCc], sre[NCc], sim[NCc];
    #pragma unroll
    for (int n = 0; n < NCc; ++n) {
        const float4 q0 = *(const float4*)(p + n * 8);
        const float4 q1 = *(const float4*)(p + n * 8 + 4);
        lre[n] = q0.x; lim[n] = q0.y; ctre[n] = q1.x; ctim[n] = q1.y;
        sre[n] = ci[2 * n]; sim[n] = ci[2 * n + 1];
    }
    float Dv = Dvec[h];
    float g  = gb[b * 2 * Hdim + h];
    float bb = gb[b * 2 * Hdim + Hdim + h];
    float rw = resw[h];
    size_t base = ((size_t)b * Ldim + (size_t)ch * CHUNK) * Hdim + h;
    #pragma unroll 2
    for (int l = 0; l < CHUNK; ++l) {
        size_t a = base + (size_t)l * Hdim;
        float uv = u[a];
        float y = Dv * uv;
        #pragma unroll
        for (int n = 0; n < NCc; ++n) {
            float r = sre[n], ii = sim[n];
            sre[n] = fmaf(lre[n], r, fmaf(-lim[n], ii, uv));
            sim[n] = fmaf(lre[n], ii, lim[n] * r);
            y = fmaf(ctre[n], sre[n], fmaf(-ctim[n], sim[n], y));
        }
        float z = gelu_fast(fmaf(y, g, bb));
        hio[a] = fmaf(hio[a], rw, z);
    }
}

// ---------------------------------------------------------------------------
// contract: out[b,l] = Σ_h h[b,l,h]*cw[h] + cb
// ---------------------------------------------------------------------------
__global__ void contract_kernel(const float* __restrict__ hbuf,
                                const float* __restrict__ cw,
                                const float* __restrict__ cb,
                                float* __restrict__ out) {
    int idx = blockIdx.x * blockDim.x + threadIdx.x;
    if (idx >= Bdim * Ldim) return;
    const float* pr = hbuf + (size_t)idx * Hdim;
    float acc = cb[0];
    #pragma unroll
    for (int h = 0; h < Hdim; h += 4) {
        float4 v = *(const float4*)(pr + h);
        float4 w = *(const float4*)(cw + h);
        acc = fmaf(v.x, w.x, acc); acc = fmaf(v.y, w.y, acc);
        acc = fmaf(v.z, w.z, acc); acc = fmaf(v.w, w.w, acc);
    }
    out[idx] = acc;
}

extern "C" void kernel_launch(void* const* d_in, const int* in_sizes, int n_in,
                              void* d_out, int out_size, void* d_ws, size_t ws_size,
                              hipStream_t stream) {
    const float* x        = (const float*)d_in[0];
    const float* params   = (const float*)d_in[1];
    const float* mlp_w1   = (const float*)d_in[2];
    const float* mlp_b1   = (const float*)d_in[3];
    const float* mlp_w2   = (const float*)d_in[4];
    const float* mlp_b2   = (const float*)d_in[5];
    const float* mlp_w3   = (const float*)d_in[6];
    const float* mlp_b3   = (const float*)d_in[7];
    const float* expand_w = (const float*)d_in[8];
    const float* expand_b = (const float*)d_in[9];
    const float* contract_w = (const float*)d_in[10];
    const float* contract_b = (const float*)d_in[11];
    const float* lin_w    = (const float*)d_in[12];
    const float* lin_b    = (const float*)d_in[13];
    const float* s4_log_dt = (const float*)d_in[14];
    const float* s4_log_A  = (const float*)d_in[15];
    const float* s4_A_imag = (const float*)d_in[16];
    const float* s4_C_re   = (const float*)d_in[17];
    const float* s4_C_im   = (const float*)d_in[18];
    const float* s4_D      = (const float*)d_in[19];
    const float* film_w    = (const float*)d_in[20];
    const float* film_b    = (const float*)d_in[21];
    const float* res_w     = (const float*)d_in[22];

    char* ws = (char*)d_ws;
    const size_t SZ_ACT = (size_t)Bdim * Ldim * Hdim * sizeof(float);        // 64 MB
    const size_t SZ_ST  = (size_t)Bdim * NCHUNK * Hdim * 16 * sizeof(float); // 8 MB
    const size_t SZ_GB  = (size_t)DEPTHc * Bdim * 2 * Hdim * sizeof(float);
    float* ws_h   = (float*)(ws);
    float* ws_t   = (float*)(ws + SZ_ACT);
    float* ws_loc = (float*)(ws + 2 * SZ_ACT);
    float* ws_car = (float*)(ws + 2 * SZ_ACT + SZ_ST);
    float* ws_gb  = (float*)(ws + 2 * SZ_ACT + 2 * SZ_ST);
    float* ws_ssm = (float*)(ws + 2 * SZ_ACT + 2 * SZ_ST + SZ_GB);

    cond_film_kernel<<<1, 256, 0, stream>>>(params, mlp_w1, mlp_b1, mlp_w2, mlp_b2,
                                            mlp_w3, mlp_b3, film_w, film_b, ws_gb);
    ssm_prep_kernel<<<(DEPTHc * Hdim * NCc + 127) / 128, 128, 0, stream>>>(
        s4_log_dt, s4_log_A, s4_A_imag, s4_C_re, s4_C_im, ws_ssm);
    expand_kernel<<<(int)(((size_t)Bdim * Ldim * Hdim) / 256), 256, 0, stream>>>(
        x, expand_w, expand_b, ws_h);

    for (int d = 0; d < DEPTHc; ++d) {
        const float* ssm_d = ws_ssm + (size_t)d * Hdim * NCc * 8;
        mix_gelu_wmma_kernel<<<dim3(Ldim / LRANGE, Bdim), 256, 0, stream>>>(
            ws_h, lin_w + (size_t)d * Hdim * Hdim, lin_b + (size_t)d * Hdim, ws_t);
        scan_phase1_kernel<<<dim3(NCHUNK, Bdim), 64, 0, stream>>>(ws_t, ssm_d, ws_loc);
        scan_phase2_kernel<<<(Bdim * Hdim + 127) / 128, 128, 0, stream>>>(ssm_d, ws_loc, ws_car);
        scan_phase3_kernel<<<dim3(NCHUNK, Bdim), 64, 0, stream>>>(
            ws_t, ws_h, ssm_d, s4_D + (size_t)d * Hdim,
            ws_gb + (size_t)d * Bdim * 2 * Hdim, res_w + (size_t)d * Hdim, ws_car);
    }

    contract_kernel<<<(Bdim * Ldim + 255) / 256, 256, 0, stream>>>(
        ws_h, contract_w, contract_b, (float*)d_out);
}
